// SetTransformerHSP_sb_7198365188348
// MI455X (gfx1250) — matrix operand order repaired and checked by reading, not compiled, not run on hardware
//
#include <hip/hip_runtime.h>
#include <hip/hip_bf16.h>

typedef __attribute__((ext_vector_type(16))) __bf16        v16bf;
typedef __attribute__((ext_vector_type(4)))  __bf16        v4bf;
typedef __attribute__((ext_vector_type(8)))  float         v8f;
typedef __attribute__((ext_vector_type(4)))  unsigned int  v4u;
typedef __attribute__((ext_vector_type(8)))  int           v8i;
typedef __attribute__((ext_vector_type(4)))  int           v4i;

// ---------------------------------------------------------------------------
// WMMA helpers (CDNA5 wave32, 16x16x32 bf16, fp32 accumulate)
// A fragment: lane holds row = lane&15; elems 0..15 -> K = kh..kh+15,
//             kh = (lane&16)?16:0.
// B fragment: lane holds col = lane&15; elems 0..15 -> K = kh..kh+15,
//             kh = (lane&16)?16:0.  (LDS staged so both reads are contiguous.)
// ---------------------------------------------------------------------------
__device__ inline v16bf frag_a(const __bf16* A, int lda, int lane) {
  const __bf16* p = A + (size_t)(lane & 15) * lda + ((lane & 16) ? 16 : 0);
  v16bf f;
#pragma unroll
  for (int i = 0; i < 16; ++i) f[i] = p[i];
  return f;
}

__device__ inline v16bf frag_b(const __bf16* Bm, int ldb, int lane) {
  const __bf16* p = Bm + (size_t)(lane & 15) * ldb + ((lane & 16) ? 16 : 0);
  v16bf f;
#pragma unroll
  for (int i = 0; i < 16; ++i) f[i] = p[i];
  return f;
}

__device__ inline v8f wmma_bf16(v16bf a, v16bf b, v8f c) {
  return __builtin_amdgcn_wmma_f32_16x16x32_bf16(false, a, false, b,
                                                 (short)0, c, false, false);
}

__device__ inline v4bf cvt4(float4 v) {
  v4bf b;
  b[0] = (__bf16)v.x; b[1] = (__bf16)v.y;
  b[2] = (__bf16)v.z; b[3] = (__bf16)v.w;
  return b;
}

// ---------------------------------------------------------------------------
// Tensor Data Mover descriptors (CDNA5 TDM, ISA ch.8 D#), bf16 2-D tiles.
// Group-1 (dims/strides) is loop-invariant and hoisted by callers; group-0
// carries the moving global address.
// ---------------------------------------------------------------------------
__device__ inline v8i tdm_group1(unsigned tensor_d0, unsigned tensor_d1,
                                 unsigned tile_d0, unsigned tile_d1,
                                 unsigned long long stride0) {
  v8i g1;
  g1[0] = (int)(1u << 16);                                // data_size = 2B
  g1[1] = (int)((tensor_d0 & 0xffffu) << 16);             // tensor_dim0 lo
  g1[2] = (int)((tensor_d0 >> 16) | ((tensor_d1 & 0xffffu) << 16));
  g1[3] = (int)((tensor_d1 >> 16) | (tile_d0 << 16));     // tile_dim0
  g1[4] = (int)(tile_d1 & 0xffffu);                       // tile_dim1
  g1[5] = (int)(stride0 & 0xffffffffull);                 // dim0 stride lo
  g1[6] = (int)((stride0 >> 32) & 0xffffull);             // dim0 stride hi
  g1[7] = 0;
  return g1;
}

__device__ inline void tdm_issue(unsigned lds_off, unsigned long long ga,
                                 v8i g1) {
  v4u g0;
  g0[0] = 1u;                                    // count=1, user descriptor
  g0[1] = lds_off;                               // LDS byte address
  g0[2] = (unsigned)(ga & 0xffffffffu);          // global_addr[31:0]
  g0[3] = (unsigned)((ga >> 32) & 0x01ffffffu)   // global_addr[56:32]
        | 0x80000000u;                           // type = 2 ("image")
  v4i z4 = {};
#if __clang_major__ >= 23
  v8i z8 = {};
  __builtin_amdgcn_tensor_load_to_lds(g0, g1, z4, z4, z8, 0);
#else
  __builtin_amdgcn_tensor_load_to_lds(g0, g1, z4, z4, 0);
#endif
}

// ---------------------------------------------------------------------------
// GEMM: out = A[M,K] @ W[N,K]^T + bias   (A,W fp32; bf16 WMMA, fp32 acc)
// block = 128 threads (4 waves); block tile 32(M) x 64(N); K step 32.
// mode 0: C fp32 row-major.   mode 1: Cb bf16 row-major.
// mode 2: Cb bf16 transposed per batch: Cb[(row/npb)*N + n][row%npb].
// Requires M%32==0, N%64==0, K%32==0 (true for every call here).
// ---------------------------------------------------------------------------
__global__ __launch_bounds__(128)
void gemm_wt_bias(const float* __restrict__ A, const float* __restrict__ W,
                  const float* __restrict__ bias,
                  float* __restrict__ C, __bf16* __restrict__ Cb,
                  int M, int N, int K, int mode, int npb) {
  __shared__ __align__(16) __bf16 As[32 * 32];
  __shared__ __align__(16) __bf16 Ws[64 * 32];
  const int tid = threadIdx.x, lane = tid & 31, wave = tid >> 5;
  const int m0 = blockIdx.y * 32;
  const int n0 = blockIdx.x * 64;
  v8f acc0 = {}, acc1 = {};
  for (int k0 = 0; k0 < K; k0 += 32) {
    // stage A (32x32) and W (64x32) as bf16.
    // Phase 1: batch all float4 loads (keeps 6 loads in flight, one wait).
    float4 la[2], lw[4];
#pragma unroll
    for (int it = 0; it < 2; ++it) {
      int e = it * 512 + tid * 4;
      la[it] = *(const float4*)(A + (size_t)(m0 + (e >> 5)) * K + k0 + (e & 31));
    }
#pragma unroll
    for (int it = 0; it < 4; ++it) {
      int e = it * 512 + tid * 4;
      lw[it] = *(const float4*)(W + (size_t)(n0 + (e >> 5)) * K + k0 + (e & 31));
    }
    if (k0 + 32 < K) {  // gfx1250 global_prefetch_b8, near-cache policy
      __builtin_prefetch(&A[(size_t)(m0 + (tid >> 2)) * K + k0 + 32], 0, 3);
      __builtin_prefetch(&W[(size_t)(n0 + (tid >> 1)) * K + k0 + 32], 0, 3);
    }
    // Phase 2: convert + LDS store
#pragma unroll
    for (int it = 0; it < 2; ++it) {
      int e = it * 512 + tid * 4;
      *(v4bf*)&As[e] = cvt4(la[it]);
    }
#pragma unroll
    for (int it = 0; it < 4; ++it) {
      int e = it * 512 + tid * 4;
      *(v4bf*)&Ws[e] = cvt4(lw[it]);
    }
    __syncthreads();
    v16bf a0 = frag_a(As, 32, lane);
    v16bf a1 = frag_a(As + 16 * 32, 32, lane);
    v16bf bf = frag_b(Ws + wave * 16 * 32, 32, lane);
    acc0 = wmma_bf16(a0, bf, acc0);
    acc1 = wmma_bf16(a1, bf, acc1);
    __syncthreads();
  }
  const int n = n0 + wave * 16 + (lane & 15);
  const float bv = bias[n];
  const int roff = (lane & 16) ? 8 : 0;
#pragma unroll
  for (int sub = 0; sub < 2; ++sub) {
    v8f acc = sub ? acc1 : acc0;
#pragma unroll
    for (int r = 0; r < 8; ++r) {
      int row = m0 + sub * 16 + roff + r;
      float val = acc[r] + bv;
      if (mode == 0) {
        C[(size_t)row * N + n] = val;
      } else if (mode == 1) {
        Cb[(size_t)row * N + n] = (__bf16)val;
      } else {
        int bz = row / npb, kv = row - bz * npb;
        Cb[((size_t)bz * N + n) * npb + kv] = (__bf16)val;
      }
    }
  }
}

// ---------------------------------------------------------------------------
// Flash-style masked multi-head attention for one MAB.
// Grid: (m/16, NUM_HEADS=4, B=16).  Block: 128 threads (4 waves).
// Kb:  bf16 (b*n, 512) row-major.      Vtb: bf16 (b, 512, n) pre-transposed.
// K/V tiles arrive via TDM (tensor_load_to_lds); adj tile is cooperatively
// staged once per block as pre-scaled mask alpha*(1-adj).
// O[b,q,h*128+d] = Qp[q] + softmax(Qh Kh^T/sqrt(512) + mask) Vh
// ---------------------------------------------------------------------------
__global__ __launch_bounds__(128)
void mab_attention(const float* __restrict__ Qp, const __bf16* __restrict__ Kb,
                   const __bf16* __restrict__ Vtb, const float* __restrict__ adj,
                   float* __restrict__ O,
                   int m, int n, int qStride, int adjRowOff, int adjColOff) {
  const int ADJ = 1088;
  const int tid = threadIdx.x, lane = tid & 31, wave = tid >> 5;
  const int q0 = blockIdx.x * 16;
  const int h  = blockIdx.y;
  const int bz = blockIdx.z;

  __shared__ __align__(16) __bf16 Qs[16 * 128];   // [q][d]
  __shared__ __align__(16) __bf16 Ks[32 * 128];   // [kv][d]  (B for S)
  __shared__ __align__(16) __bf16 Vts[128 * 32];  // [d][kv]  (B for O)
  __shared__ __align__(16) __bf16 Ps[16 * 32];    // [q][kv]  (A for O)
  __shared__ __align__(16) float  Ms[16 * 32];    // alpha*(1-adj) tile

  const size_t qbase = (size_t)bz * qStride + (size_t)q0 * 512 + h * 128;
  {
    float4 lq[4];
#pragma unroll
    for (int it = 0; it < 4; ++it) {   // Q tile: 16x128 floats
      int e = it * 512 + tid * 4;
      lq[it] = *(const float4*)(Qp + qbase + (size_t)(e >> 7) * 512 + (e & 127));
    }
#pragma unroll
    for (int it = 0; it < 4; ++it) {
      int e = it * 512 + tid * 4;
      *(v4bf*)&Qs[e] = cvt4(lq[it]);
    }
  }

  const float alpha = (h == 0) ? 0.0f : (h == 1) ? -1.0f
                     : (h == 2) ? -5.0f : -1.0e9f;
  const float scale = 0.04419417382415922f;  // 1/sqrt(512)
  const int qloc = (lane & 16) ? 8 : 0;
  const int col  = lane & 15;

  // loop-invariant TDM descriptors; only global address moves per iteration
  const v8i g1K = tdm_group1(128, 32, 128, 32, 512ull);
  const v8i g1V = tdm_group1(32, 128, 32, 128, (unsigned long long)n);
  const unsigned ldsK = (unsigned)(uintptr_t)&Ks[0];
  const unsigned ldsV = (unsigned)(uintptr_t)&Vts[0];
  unsigned long long gaK =
      (unsigned long long)(uintptr_t)(Kb + ((size_t)bz * n) * 512 + h * 128);
  unsigned long long gaV =
      (unsigned long long)(uintptr_t)(Vtb + ((size_t)bz * 512 + h * 128) * n);
  const float* adjRow0 =
      adj + (size_t)bz * ADJ * ADJ + (size_t)(adjRowOff + q0) * ADJ + adjColOff;

  float mrow[8], lrow[8];
  v8f oacc0 = {}, oacc1 = {};
#pragma unroll
  for (int r = 0; r < 8; ++r) { mrow[r] = -__builtin_inff(); lrow[r] = 0.0f; }
  __syncthreads();

  for (int kv0 = 0; kv0 < n; kv0 += 32) {
    // ---- TDM: DMA K (32x128) and V^T (128x32) bf16 tiles into LDS ----
    if (wave == 0) {
      tdm_issue(ldsK, gaK, g1K);
      tdm_issue(ldsV, gaV, g1V);
      __builtin_amdgcn_s_wait_tensorcnt(0);
    }
    gaK += 32ull * 512ull * 2ull;
    gaV += 32ull * 2ull;
    // ---- cooperative mask tile: Ms[q][kv] = alpha*(1-adj), 1 float4/thr ----
    if (h != 0) {
      int e = tid * 4;
      int r = e >> 5, c = e & 31;
      float4 v = *(const float4*)(adjRow0 + (size_t)r * ADJ + kv0 + c);
      float4 mm;
      mm.x = alpha * (1.0f - v.x); mm.y = alpha * (1.0f - v.y);
      mm.z = alpha * (1.0f - v.z); mm.w = alpha * (1.0f - v.w);
      *(float4*)&Ms[e] = mm;
    }
    __syncthreads();

    // ---- S = Q @ K^T  (16x32 scores; two 16-col n-tiles; K-dim 128) ----
    v8f s0 = {}, s1 = {};
#pragma unroll
    for (int kc = 0; kc < 4; ++kc) {
      v16bf af = frag_a(Qs + kc * 32, 128, lane);
      v16bf b0 = frag_b(Ks + kc * 32, 128, lane);
      v16bf b1 = frag_b(Ks + 16 * 128 + kc * 32, 128, lane);
      s0 = wmma_bf16(af, b0, s0);
      s1 = wmma_bf16(af, b1, s1);
    }

    // ---- mask + online softmax (row stats via 16-lane xor shuffles) ----
    float p0[8], p1[8];
#pragma unroll
    for (int r = 0; r < 8; ++r) {
      float v0 = s0[r] * scale, v1 = s1[r] * scale;
      if (h != 0) {
        v0 += Ms[(qloc + r) * 32 + col];
        v1 += Ms[(qloc + r) * 32 + 16 + col];
      }
      float t = fmaxf(v0, v1);
      t = fmaxf(t, __shfl_xor(t, 1));
      t = fmaxf(t, __shfl_xor(t, 2));
      t = fmaxf(t, __shfl_xor(t, 4));
      t = fmaxf(t, __shfl_xor(t, 8));
      float mn = fmaxf(mrow[r], t);
      float corr = __expf(mrow[r] - mn);
      mrow[r] = mn;
      float e0 = __expf(v0 - mn), e1 = __expf(v1 - mn);
      float rs = e0 + e1;
      rs += __shfl_xor(rs, 1);
      rs += __shfl_xor(rs, 2);
      rs += __shfl_xor(rs, 4);
      rs += __shfl_xor(rs, 8);
      lrow[r] = lrow[r] * corr + rs;
      oacc0[r] *= corr;
      oacc1[r] *= corr;
      p0[r] = e0; p1[r] = e1;
    }

    // ---- P -> LDS in A-fragment layout (all waves hold identical P) ----
    if (wave == 0) {
#pragma unroll
      for (int r = 0; r < 8; ++r) {
        Ps[(qloc + r) * 32 + col]      = (__bf16)p0[r];
        Ps[(qloc + r) * 32 + 16 + col] = (__bf16)p1[r];
      }
    }
    __syncthreads();

    // ---- O += P @ V   (each wave: 2 of the 8 d-tiles) ----
    v16bf pa  = frag_a(Ps, 32, lane);
    v16bf vb0 = frag_b(Vts + (wave * 2 + 0) * 16 * 32, 32, lane);
    v16bf vb1 = frag_b(Vts + (wave * 2 + 1) * 16 * 32, 32, lane);
    oacc0 = wmma_bf16(pa, vb0, oacc0);
    oacc1 = wmma_bf16(pa, vb1, oacc1);
    __syncthreads();
  }

  // ---- epilogue: normalize, add Q residual, store ----
#pragma unroll
  for (int r = 0; r < 8; ++r) {
    int q = qloc + r;
    float inv = 1.0f / lrow[r];
    int d0 = (wave * 2 + 0) * 16 + col;
    int d1 = (wave * 2 + 1) * 16 + col;
    size_t qrow = qbase + (size_t)q * 512;
    size_t orow = ((size_t)bz * m + q0 + q) * 512 + h * 128;
    O[orow + d0] = Qp[qrow + d0] + oacc0[r] * inv;
    O[orow + d1] = Qp[qrow + d1] + oacc1[r] * inv;
  }
}

// ---------------------------------------------------------------------------
// LayerNorm over last dim (512), optional fused  x := x + relu(R)  pre-add.
// ---------------------------------------------------------------------------
__global__ __launch_bounds__(256)
void ln_kernel(const float* __restrict__ X, const float* __restrict__ R,
               const float* __restrict__ g, const float* __restrict__ bt,
               float* __restrict__ Y) {
  __shared__ float s1[256], s2[256];
  const int tid = threadIdx.x;
  const size_t base = (size_t)blockIdx.x * 512;
  float x0 = X[base + tid];
  float x1 = X[base + 256 + tid];
  if (R) {
    x0 += fmaxf(R[base + tid], 0.0f);
    x1 += fmaxf(R[base + 256 + tid], 0.0f);
  }
  s1[tid] = x0 + x1;
  s2[tid] = x0 * x0 + x1 * x1;
  __syncthreads();
  for (int off = 128; off > 0; off >>= 1) {
    if (tid < off) { s1[tid] += s1[tid + off]; s2[tid] += s2[tid + off]; }
    __syncthreads();
  }
  const float mean = s1[0] * (1.0f / 512.0f);
  const float var  = s2[0] * (1.0f / 512.0f) - mean * mean;
  const float inv  = rsqrtf(var + 1e-5f);
  Y[base + tid]       = (x0 - mean) * inv * g[tid]       + bt[tid];
  Y[base + 256 + tid] = (x1 - mean) * inv * g[256 + tid] + bt[256 + tid];
}

// ---------------------------------------------------------------------------
// Host orchestration
// ---------------------------------------------------------------------------
struct MabP {
  const float *wq, *bq, *wk, *bk, *wv, *bv, *wo, *bo, *g0, *b0, *g1, *b1;
};

extern "C" void kernel_launch(void* const* d_in, const int* in_sizes, int n_in,
                              void* d_out, int out_size, void* d_ws, size_t ws_size,
                              hipStream_t stream) {
  (void)in_sizes; (void)n_in; (void)out_size; (void)ws_size;
  const int Bb = 16, V = 1024, DH = 512, HS = 64, DIN = 192;

  const float* x   = (const float*)d_in[0];   // (16,1024,12,16) -> (16384,192)
  const float* adj = (const float*)d_in[1];   // (16,1088,1088)

  MabP mp[5];
  for (int l = 0; l < 5; ++l) {
    int b0 = 2 + 12 * l;
    mp[l].wq = (const float*)d_in[b0 + 0];  mp[l].bq = (const float*)d_in[b0 + 1];
    mp[l].wk = (const float*)d_in[b0 + 2];  mp[l].bk = (const float*)d_in[b0 + 3];
    mp[l].wv = (const float*)d_in[b0 + 4];  mp[l].bv = (const float*)d_in[b0 + 5];
    mp[l].wo = (const float*)d_in[b0 + 6];  mp[l].bo = (const float*)d_in[b0 + 7];
    mp[l].g0 = (const float*)d_in[b0 + 8];  mp[l].b0 = (const float*)d_in[b0 + 9];
    mp[l].g1 = (const float*)d_in[b0 +10];  mp[l].b1 = (const float*)d_in[b0 +11];
  }
  const float* Sparam = (const float*)d_in[62];  // (64,512) broadcast
  const float* w_out  = (const float*)d_in[63];  // (192,512)
  const float* b_out  = (const float*)d_in[64];  // (192)

  // ---- scratch layout ----
  const size_t BIG = (size_t)Bb * V * DH;   // 16384*512 elements
  const size_t SML = (size_t)Bb * HS * DH;  // 1024*512 elements
  char* wsb = (char*)d_ws;
  float*  bufE = (float*)wsb;                      wsb += BIG * 4;
  float*  bufQ = (float*)wsb;                      wsb += BIG * 4;
  float*  bufK = (float*)wsb;                      wsb += BIG * 4;
  __bf16* Kbf  = (__bf16*)wsb;                     wsb += BIG * 2;
  __bf16* Vtbf = (__bf16*)wsb;                     wsb += BIG * 2;
  float*  oE   = (float*)wsb;                      wsb += SML * 4;
  float*  oQ   = (float*)wsb;                      wsb += SML * 4;
  float*  oK   = (float*)wsb;                      wsb += SML * 4;
  __bf16* oKb  = (__bf16*)wsb;                     wsb += SML * 2;
  __bf16* oVb  = (__bf16*)wsb;                     wsb += SML * 2;

  auto gemm = [&](const float* A, const float* Wm, const float* bias,
                  float* C, __bf16* Cb, int M, int N, int K, int mode, int npb) {
    dim3 grid(N / 64, M / 32);
    gemm_wt_bias<<<grid, dim3(128), 0, stream>>>(A, Wm, bias, C, Cb,
                                                 M, N, K, mode, npb);
  };
  auto attn = [&](const float* Qp, const __bf16* Kb, const __bf16* Vtb,
                  float* O, int m, int n, int qStride, int rowOff, int colOff) {
    dim3 grid(m / 16, 4, Bb);
    mab_attention<<<grid, dim3(128), 0, stream>>>(Qp, Kb, Vtb, adj, O,
                                                  m, n, qStride, rowOff, colOff);
  };
  auto ln = [&](const float* X, const float* R, const float* g, const float* bt,
                float* Y, int rows) {
    ln_kernel<<<dim3(rows), dim3(256), 0, stream>>>(X, R, g, bt, Y);
  };

  // ---- enc0: MAB(xf, xf, adj[:V,:V]) -> bufE ----
  gemm(x, mp[0].wq, mp[0].bq, bufQ, nullptr, Bb * V, DH, DIN, 0, 0);
  gemm(x, mp[0].wk, mp[0].bk, nullptr, Kbf,  Bb * V, DH, DIN, 1, 0);
  gemm(x, mp[0].wv, mp[0].bv, nullptr, Vtbf, Bb * V, DH, DIN, 2, V);
  attn(bufQ, Kbf, Vtbf, bufE, V, V, V * DH, 0, 0);
  ln(bufE, nullptr, mp[0].g0, mp[0].b0, bufK, Bb * V);            // O1 -> bufK
  gemm(bufK, mp[0].wo, mp[0].bo, bufQ, nullptr, Bb * V, DH, DH, 0, 0);  // R
  ln(bufK, bufQ, mp[0].g1, mp[0].b1, bufE, Bb * V);               // out -> bufE

  // ---- enc1: MAB(e, e, adj[:V,:V]) -> bufE ----
  gemm(bufE, mp[1].wq, mp[1].bq, bufQ, nullptr, Bb * V, DH, DH, 0, 0);
  gemm(bufE, mp[1].wk, mp[1].bk, nullptr, Kbf,  Bb * V, DH, DH, 1, 0);
  gemm(bufE, mp[1].wv, mp[1].bv, nullptr, Vtbf, Bb * V, DH, DH, 2, V);
  attn(bufQ, Kbf, Vtbf, bufE, V, V, V * DH, 0, 0);
  ln(bufE, nullptr, mp[1].g0, mp[1].b0, bufK, Bb * V);
  gemm(bufK, mp[1].wo, mp[1].bo, bufQ, nullptr, Bb * V, DH, DH, 0, 0);
  ln(bufK, bufQ, mp[1].g1, mp[1].b1, bufE, Bb * V);

  // ---- pma: MAB(S, e, adj[V:, :V]) -> oE ----
  gemm(Sparam, mp[2].wq, mp[2].bq, oQ, nullptr, HS, DH, DH, 0, 0);  // 64 rows
  gemm(bufE, mp[2].wk, mp[2].bk, nullptr, Kbf,  Bb * V, DH, DH, 1, 0);
  gemm(bufE, mp[2].wv, mp[2].bv, nullptr, Vtbf, Bb * V, DH, DH, 2, V);
  attn(oQ, Kbf, Vtbf, oE, HS, V, 0, V, 0);
  ln(oE, nullptr, mp[2].g0, mp[2].b0, oK, Bb * HS);
  gemm(oK, mp[2].wo, mp[2].bo, oQ, nullptr, Bb * HS, DH, DH, 0, 0);
  ln(oK, oQ, mp[2].g1, mp[2].b1, oE, Bb * HS);

  // ---- dec1, dec2: MAB(o, o, adj[V:, V:]) -> oE ----
  for (int l = 3; l <= 4; ++l) {
    gemm(oE, mp[l].wq, mp[l].bq, oQ, nullptr, Bb * HS, DH, DH, 0, 0);
    gemm(oE, mp[l].wk, mp[l].bk, nullptr, oKb, Bb * HS, DH, DH, 1, 0);
    gemm(oE, mp[l].wv, mp[l].bv, nullptr, oVb, Bb * HS, DH, DH, 2, HS);
    attn(oQ, oKb, oVb, oE, HS, HS, HS * DH, V, V);
    ln(oE, nullptr, mp[l].g0, mp[l].b0, oK, Bb * HS);
    gemm(oK, mp[l].wo, mp[l].bo, oQ, nullptr, Bb * HS, DH, DH, 0, 0);
    ln(oK, oQ, mp[l].g1, mp[l].b1, oE, Bb * HS);
  }

  // ---- output projection: (1024,512) @ (192,512)^T + b -> d_out ----
  gemm(oE, w_out, b_out, (float*)d_out, nullptr, Bb * HS, 192, DH, 0, 0);
}
